// LiquidNeuralNetworkMultiStep_27504970563771
// MI455X (gfx1250) — compile-verified, matrix operand order checked by reading, not run on hardware
//
#include <hip/hip_runtime.h>
#include <hip/hip_bf16.h>

typedef __bf16 bf16_t;
typedef __attribute__((ext_vector_type(16))) __bf16 v16bf;
typedef __attribute__((ext_vector_type(8)))  __bf16 v8bf;
typedef __attribute__((ext_vector_type(8)))  float  v8f;
typedef __attribute__((ext_vector_type(4)))  unsigned int v4u;
typedef __attribute__((ext_vector_type(8)))  int  v8i;
typedef __attribute__((ext_vector_type(4)))  int  v4i;

#define HDIM     512
#define FDIM     8
#define TSTEPS   50
#define POUT     10
#define BM       32            // batch rows per workgroup
#define LDSTRIDE 520           // 512 + 8 halves pad -> conflict-optimal A-frag b128 LDS loads
#define WROWH    40            // weight-tile LDS row stride in halves (64B data + 16B pad)
#define NTILES   48            // 16 (Wh0) + 16 (Wh1) + 16 (Wu1) K-tiles per step
#define DT_C     0.1f

__device__ __forceinline__ float fast_tanh(float v) {
#if __has_builtin(__builtin_amdgcn_tanh_f32)
    return __builtin_amdgcn_tanh_f32(v);
#else
    return tanhf(v);
#endif
}

// 16x32 bf16 A/B fragment (per-lane chunks at K+8g and K+16+8g), ISA 7.12.2 layout.
__device__ __forceinline__ v16bf load_frag(const bf16_t* p, int g) {
    const v8bf lo = *(const v8bf*)(p + 8 * g);
    const v8bf hi = *(const v8bf*)(p + 16 + 8 * g);
    return __builtin_shufflevector(lo, hi, 0,1,2,3,4,5,6,7,8,9,10,11,12,13,14,15);
}

// ---- Tensor Data Mover: DMA one 32K x 512N bf16 weight tile (64B rows,
// 1024B tensor stride) into LDS with 16B padding per row (80B LDS stride). ----
__device__ __forceinline__ void tdm_issue(const bf16_t* gsrc, unsigned lds_off) {
#if __has_builtin(__builtin_amdgcn_tensor_load_to_lds)
    unsigned long long ga = (unsigned long long)(uintptr_t)gsrc;
    v4u g0;
    g0[0] = 1u;                                   // count=1 (valid user descriptor)
    g0[1] = lds_off;                              // lds_addr (bytes)
    g0[2] = (unsigned)ga;                         // global_addr[31:0]
    g0[3] = (unsigned)((ga >> 32) & 0x01FFFFFFull) | (2u << 30);  // addr[56:32] | type=2
    v8i g1;
    g1[0] = (int)((3u << 16)      // data_size = 8B units
                | (1u << 20)      // pad_enable
                | (3u << 22)      // pad_interval: 16 DWORDs (=64B row)
                | (3u << 25));    // pad_amount: 4 DWORDs (=16B) -> 80B LDS row stride
    g1[1] = (int)(8u << 16);      // tensor_dim0 = 8 units (64B row)
    g1[2] = (int)(512u << 16);    // tensor_dim1 = 512 rows
    g1[3] = (int)(8u << 16);      // tile_dim0 = 8 units
    g1[4] = (int)512u;            // tile_dim1 = 512 (tile_dim2 = 0)
    g1[5] = (int)128u;            // tensor_dim0_stride = 128 units (1024B)
    g1[6] = 0;
    g1[7] = 0;
    v4i z4 = {0, 0, 0, 0};
#if defined(__clang_major__) && __clang_major__ >= 23
    v8i z8 = {0, 0, 0, 0, 0, 0, 0, 0};
    __builtin_amdgcn_tensor_load_to_lds(g0, g1, z4, z4, z8, 0);
#else
    __builtin_amdgcn_tensor_load_to_lds(g0, g1, z4, z4, 0);
#endif
#else
    // Fallback: synchronous cooperative copy by the issuing wave (compile safety).
    extern __shared__ bf16_t smem_base[];
    bf16_t* dst = (bf16_t*)((char*)smem_base + lds_off);
    const int l = threadIdx.x & 31;
    for (int i = l; i < 512 * 8; i += 32) {
        const int n = i >> 3, q = i & 7;
        ((unsigned long long*)(dst + n * WROWH))[q] =
            ((const unsigned long long*)(gsrc + (size_t)n * HDIM))[q];
    }
#endif
}

__device__ __forceinline__ void tdm_wait1() {
#if __has_builtin(__builtin_amdgcn_s_wait_tensorcnt)
    __builtin_amdgcn_s_wait_tensorcnt((short)1);
#endif
}

// f32 [K][N] row-major -> bf16 [N][Kp] row-major (transposed, zero-padded K)
__global__ void lnn_convert_transpose(const float* __restrict__ W, bf16_t* __restrict__ WT,
                                      int K, int N, int Kp) {
    int idx = blockIdx.x * blockDim.x + threadIdx.x;
    if (idx >= N * Kp) return;
    int n = idx / Kp;
    int k = idx - n * Kp;
    WT[idx] = (k < K) ? (bf16_t)W[k * N + n] : (bf16_t)0.0f;
}

__global__ __launch_bounds__(256, 1)
void lnn_persistent(const float* __restrict__ x,
                    const float* __restrict__ b0,
                    const float* __restrict__ b1,
                    const float* __restrict__ fc_w,
                    const float* __restrict__ fc_b,
                    const bf16_t* __restrict__ Wh0T,
                    const bf16_t* __restrict__ Wu0Tp,
                    const bf16_t* __restrict__ Wh1T,
                    const bf16_t* __restrict__ Wu1T,
                    float* __restrict__ out) {
    __shared__ bf16_t hA0[BM * LDSTRIDE];       // bf16 state h0 (A operand)
    __shared__ bf16_t hA1[BM * LDSTRIDE];       // bf16 state h1
    __shared__ bf16_t wtile[2][HDIM * WROWH];   // double-buffered TDM weight tiles

    const int tid    = threadIdx.x;
    const int wave   = tid >> 5;
    const int lane   = tid & 31;
    const int g      = lane >> 4;
    const int ln     = lane & 15;
    const int m_base = (wave & 1) * 16;         // 2 M-tiles
    const int n_base = (wave >> 1) * 128;       // 4 N-blocks x 8 WMMA tiles
    const int rb     = blockIdx.x * BM;

    const unsigned wt_off0 = (unsigned)(uintptr_t)&wtile[0][0];
    const unsigned wt_off1 = (unsigned)(uintptr_t)&wtile[1][0];

    for (int i = tid; i < BM * LDSTRIDE; i += 256) {
        hA0[i] = (bf16_t)0.0f;
        hA1[i] = (bf16_t)0.0f;
    }
    __syncthreads();

    const int rowA = m_base + ln;

    for (int t = 0; t < TSTEPS; ++t) {
        v8f acc[8];
        int s = 0;   // tile sequence: 0..15 Wh0 | 16..31 Wh1 | 32..47 Wu1

        // kick the TDM pipeline for this step: tiles 0 and 1
        if (wave == 0) {
            tdm_issue(Wh0T, wt_off0);
            tdm_issue(Wh0T + 32, wt_off1);
        }

        // ===== layer 0 init: b0 + u_t@Wu0 (one zero-padded K=32 WMMA) =====
        #pragma unroll
        for (int j = 0; j < 8; ++j) {
            const float bias = b0[n_base + j * 16 + ln];
            #pragma unroll
            for (int e = 0; e < 8; ++e) acc[j][e] = bias;
        }
        {
            v16bf ax;
            #pragma unroll
            for (int i = 0; i < 16; ++i) ax[i] = (bf16_t)0.0f;
            if (g == 0) {
                const float* xp = x + (size_t)(rb + rowA) * (TSTEPS * FDIM) + t * FDIM;
                #pragma unroll
                for (int i = 0; i < 8; ++i) ax[i] = (bf16_t)xp[i];
            }
            #pragma unroll
            for (int j = 0; j < 8; ++j) {
                const int n = n_base + j * 16 + ln;
                const v16bf bx = load_frag(Wu0Tp + (size_t)n * 32, g);
                acc[j] = __builtin_amdgcn_wmma_f32_16x16x32_bf16(
                    false, ax, false, bx, (short)0, acc[j], false, false);
            }
        }
        if (wave == 0) tdm_wait1();   // tile 0 landed
        __syncthreads();

        // ===== pass A: h0 @ Wh0 =====
        #pragma unroll 1
        for (int kt = 0; kt < HDIM / 32; ++kt, ++s) {
            const v16bf a = load_frag(&hA0[rowA * LDSTRIDE + kt * 32], g);
            const bf16_t* wt = &wtile[s & 1][0];
            #pragma unroll
            for (int j = 0; j < 8; ++j) {
                const int n = n_base + j * 16 + ln;
                const v16bf b = load_frag(wt + n * WROWH, g);
                acc[j] = __builtin_amdgcn_wmma_f32_16x16x32_bf16(
                    false, a, false, b, (short)0, acc[j], false, false);
            }
            __syncthreads();                         // all waves done with buf s&1
            if (wave == 0) {
                if (s + 2 < NTILES) {
                    const int s2 = s + 2;
                    const bf16_t* src = (s2 < 16) ? (Wh0T + (size_t)s2 * 32)
                                      : (s2 < 32) ? (Wh1T + (size_t)(s2 - 16) * 32)
                                                  : (Wu1T + (size_t)(s2 - 32) * 32);
                    tdm_issue(src, (s & 1) ? wt_off1 : wt_off0);
                }
                tdm_wait1();                         // tile s+1 landed
            }
            __syncthreads();
        }

        // h0 <- (1-dt)h0 + dt*tanh(G0)
        #pragma unroll 1
        for (int j = 0; j < 8; ++j) {
            #pragma unroll
            for (int e = 0; e < 8; ++e) {
                const int off = (m_base + e + 8 * g) * LDSTRIDE + n_base + j * 16 + ln;
                const float hold = (float)hA0[off];
                hA0[off] = (bf16_t)((1.0f - DT_C) * hold + DT_C * fast_tanh(acc[j][e]));
            }
        }
        __syncthreads();

        // ===== layer 1 init: bias =====
        #pragma unroll
        for (int j = 0; j < 8; ++j) {
            const float bias = b1[n_base + j * 16 + ln];
            #pragma unroll
            for (int e = 0; e < 8; ++e) acc[j][e] = bias;
        }
        // ===== pass B: h1 @ Wh1 ; pass C: h0 @ Wu1 (same accumulators) =====
        #pragma unroll 1
        for (int p = 0; p < 2; ++p) {
            const bf16_t* hA = (p == 0) ? hA1 : hA0;
            #pragma unroll 1
            for (int kt = 0; kt < HDIM / 32; ++kt, ++s) {
                const v16bf a = load_frag(&hA[rowA * LDSTRIDE + kt * 32], g);
                const bf16_t* wt = &wtile[s & 1][0];
                #pragma unroll
                for (int j = 0; j < 8; ++j) {
                    const int n = n_base + j * 16 + ln;
                    const v16bf b = load_frag(wt + n * WROWH, g);
                    acc[j] = __builtin_amdgcn_wmma_f32_16x16x32_bf16(
                        false, a, false, b, (short)0, acc[j], false, false);
                }
                __syncthreads();
                if (wave == 0) {
                    if (s + 2 < NTILES) {
                        const int s2 = s + 2;
                        const bf16_t* src = (s2 < 16) ? (Wh0T + (size_t)s2 * 32)
                                          : (s2 < 32) ? (Wh1T + (size_t)(s2 - 16) * 32)
                                                      : (Wu1T + (size_t)(s2 - 32) * 32);
                        tdm_issue(src, (s & 1) ? wt_off1 : wt_off0);
                    }
                    tdm_wait1();
                }
                __syncthreads();
            }
        }

        // h1 <- (1-dt)h1 + dt*tanh(G1)
        #pragma unroll 1
        for (int j = 0; j < 8; ++j) {
            #pragma unroll
            for (int e = 0; e < 8; ++e) {
                const int off = (m_base + e + 8 * g) * LDSTRIDE + n_base + j * 16 + ln;
                const float hold = (float)hA1[off];
                hA1[off] = (bf16_t)((1.0f - DT_C) * hold + DT_C * fast_tanh(acc[j][e]));
            }
        }
        __syncthreads();
    }

    // ===== head: out = h1 @ fc_w + fc_b (512x10, VALU) =====
    for (int idx = tid; idx < BM * POUT; idx += 256) {
        const int r = idx / POUT;
        const int p = idx - r * POUT;
        float sum = fc_b[p];
        #pragma unroll 1
        for (int k = 0; k < HDIM; ++k)
            sum += (float)hA1[r * LDSTRIDE + k] * fc_w[k * POUT + p];
        out[(size_t)(rb + r) * POUT + p] = sum;
    }
}

extern "C" void kernel_launch(void* const* d_in, const int* in_sizes, int n_in,
                              void* d_out, int out_size, void* d_ws, size_t ws_size,
                              hipStream_t stream) {
    const float* x   = (const float*)d_in[0];
    const float* Wh0 = (const float*)d_in[1];
    const float* Wu0 = (const float*)d_in[2];
    const float* b0  = (const float*)d_in[3];
    const float* Wh1 = (const float*)d_in[4];
    const float* Wu1 = (const float*)d_in[5];
    const float* b1  = (const float*)d_in[6];
    const float* fcw = (const float*)d_in[7];
    const float* fcb = (const float*)d_in[8];
    float* out = (float*)d_out;

    const int B = in_sizes[0] / (TSTEPS * FDIM);   // 8192

    bf16_t* Wh0T  = (bf16_t*)d_ws;                 // bf16 [N][K] transposed weights
    bf16_t* Wh1T  = Wh0T + HDIM * HDIM;
    bf16_t* Wu1T  = Wh1T + HDIM * HDIM;
    bf16_t* Wu0Tp = Wu1T + HDIM * HDIM;            // [512][32], K padded 8->32

    const int thr = 256;
    lnn_convert_transpose<<<(HDIM * HDIM + thr - 1) / thr, thr, 0, stream>>>(Wh0, Wh0T, HDIM, HDIM, HDIM);
    lnn_convert_transpose<<<(HDIM * HDIM + thr - 1) / thr, thr, 0, stream>>>(Wh1, Wh1T, HDIM, HDIM, HDIM);
    lnn_convert_transpose<<<(HDIM * HDIM + thr - 1) / thr, thr, 0, stream>>>(Wu1, Wu1T, HDIM, HDIM, HDIM);
    lnn_convert_transpose<<<(HDIM * 32  + thr - 1) / thr, thr, 0, stream>>>(Wu0, Wu0Tp, FDIM, HDIM, 32);

    lnn_persistent<<<B / BM, 256, 0, stream>>>(x, b0, b1, fcw, fcb,
                                               Wh0T, Wu0Tp, Wh1T, Wu1T, out);
}
